// Discriminator_75282186764602
// MI455X (gfx1250) — matrix operand order repaired and checked by reading, not compiled, not run on hardware
//
#include <hip/hip_runtime.h>
#include <hip/hip_bf16.h>
#include <stdint.h>

// ---------------------------------------------------------------------------
// Discriminator forward for MI455X (gfx1250, wave32).
// conv2 (the 105-GFLOP implicit GEMM) runs on v_wmma_f32_16x16x32_f16 with
// double-buffered GLOBAL_LOAD_ASYNC_TO_LDS_B128 tile staging (ASYNCcnt).
// ---------------------------------------------------------------------------

typedef __attribute__((ext_vector_type(16))) _Float16 v16h;
typedef __attribute__((ext_vector_type(8)))  float    v8f;
typedef __attribute__((ext_vector_type(4)))  int      v4i;

#define B_SZ      8192
#define M_ROWS    (B_SZ * 49)          // 401408 output pixels of conv2
#define K_DIM     1024                 // 64 ic * 16 taps, ordered (kh,kw,ic)
#define N_DIM     128
#define BN_EPS    1e-5f

// ---- CDNA5 async memory->LDS staging (guarded; falls back to plain copy) --
#if defined(__has_builtin)
#if __has_builtin(__builtin_amdgcn_global_load_async_to_lds_b128) && \
    __has_builtin(__builtin_amdgcn_s_wait_asynccnt)
#define USE_ASYNC_LDS 1
#endif
#endif
#ifndef USE_ASYNC_LDS
#define USE_ASYNC_LDS 0
#endif

#if USE_ASYNC_LDS
typedef __attribute__((address_space(1))) v4i gv4i;   // global int4
typedef __attribute__((address_space(3))) v4i lv4i;   // LDS int4
static __device__ __forceinline__ void async_copy_b128(void* lds_dst, const void* gsrc) {
    // (global src, LDS dst, imm offset, imm cpol)
    __builtin_amdgcn_global_load_async_to_lds_b128((gv4i*)gsrc, (lv4i*)lds_dst, 0, 0);
}
#endif

static __device__ __forceinline__ float leaky(float x) {
    return x >= 0.f ? x : 0.2f * x;
}

// ---------------------------------------------------------------------------
// Kernel 0: zero the BN-stats accumulators (gsum|gsq contiguous, 256 floats)
// ---------------------------------------------------------------------------
__global__ void zero_f32(float* __restrict__ p, int n) {
    int i = blockIdx.x * 256 + threadIdx.x;
    if (i < n) p[i] = 0.f;
}

// ---------------------------------------------------------------------------
// Kernel 1: conv1 (one-hot trick: only channel label_b is live) + leaky.
// Output h1 in NHWC f16: h1[b][oy][ox][oc], oc fastest -> conv2 A-tile rows
// are contiguous 64-channel segments.
// ---------------------------------------------------------------------------
__global__ __launch_bounds__(256) void conv1_leaky(
    const float* __restrict__ img,      // [B][1][28][28]
    const int*   __restrict__ labels,   // [B]
    const float* __restrict__ w1,       // [64][10][4][4]
    _Float16*    __restrict__ h1)       // [B][14][14][64]
{
    __shared__ float simg[28 * 28];
    __shared__ float sw[64 * 16];       // w1[oc][label][kh][kw]

    const int b   = blockIdx.x;
    const int t   = threadIdx.x;
    const int lab = labels[b];

    for (int i = t; i < 784; i += 256) simg[i] = img[(size_t)b * 784 + i];
    for (int i = t; i < 1024; i += 256) {
        int oc = i >> 4, k = i & 15;
        sw[i] = w1[((oc * 10 + lab) << 4) + k];
    }
    __syncthreads();

    for (int j = t; j < 12544; j += 256) {          // 196 pixels * 64 oc
        int oc  = j & 63;
        int pix = j >> 6;
        int oy  = pix / 14, ox = pix % 14;
        float acc = 0.f;
#pragma unroll
        for (int kh = 0; kh < 4; ++kh) {
            int iy = 2 * oy - 1 + kh;
            if (iy < 0 || iy >= 28) continue;
#pragma unroll
            for (int kw = 0; kw < 4; ++kw) {
                int ix = 2 * ox - 1 + kw;
                if (ix < 0 || ix >= 28) continue;
                acc += simg[iy * 28 + ix] * sw[(oc << 4) + kh * 4 + kw];
            }
        }
        h1[(size_t)b * 12544 + (size_t)pix * 64 + oc] = (_Float16)leaky(acc);
    }
}

// ---------------------------------------------------------------------------
// Kernel 2: repack conv2 weights OIHW f32 -> w2t[n=oc][k=(kh*4+kw)*64+ic] f16
// ---------------------------------------------------------------------------
__global__ __launch_bounds__(256) void prep_w2(
    const float* __restrict__ w2,       // [128][64][4][4]
    _Float16*    __restrict__ w2t)      // [128][1024]
{
    int idx = blockIdx.x * 256 + threadIdx.x;       // 131072 total
    if (idx >= N_DIM * K_DIM) return;
    int oc = idx >> 10;
    int k  = idx & 1023;
    int khkw = k >> 6, ic = k & 63;
    int kh = khkw >> 2, kw = khkw & 3;
    w2t[idx] = (_Float16)w2[((oc * 64 + ic) * 4 + kh) * 4 + kw];
}

// ---------------------------------------------------------------------------
// Kernel 3: conv2 as implicit GEMM on WMMA.
//   M = B*49 rows (output pixel), N = 128 oc, K = 1024, f16 x f16 -> f32.
//   Block: 128 threads (4 waves). M-tile 64 (16 rows/wave), full N per wave.
//   K-tile 64 = exactly one (kh,kw) tap -> A row slices are contiguous 128 B
//   of NHWC h1, staged with global_load_async_to_lds_b128 (ASYNCcnt),
//   double-buffered so tap kc+1 streams in while tap kc is computed
//   (one barrier per iteration; wait_asynccnt has a full compute phase to hide).
//   LDS row stride 72 halves (144 B): 16B-aligned for B128 async stores and
//   a 36-dword pitch so the 16 row-gather lanes hit 16 distinct banks.
// Fragment layouts follow CDNA5 ISA §7.12.2:
//   A 16x32 f16: lanes 0-15 = M, VGPR r<4: K = g*8+2r+h, r>=4: K = 16+g*8+2(r-4)+h
//   B 32x16 f16: same K order as A (lanes = N)
//   C/D 16x16 f32: VGPR r: M = r + 8*g, N = lane%16
// ---------------------------------------------------------------------------
#define LSTR 72                          // LDS row stride in halves

// Stage one tap's A (64x64) and B (128x64) tiles into LDS buffers.
static __device__ __forceinline__ void stage_conv2_tiles(
    _Float16* __restrict__ Asb,          // A buffer base (this double-buffer slot)
    _Float16* __restrict__ Bsb,          // B buffer base
    int kc, int tid, int arow, int aseg, int oy, int ox,
    const _Float16* __restrict__ h1b,    // h1 + b_*196*64 (this row's sample)
    const _Float16* __restrict__ w2row)  // w2t + tid*K_DIM
{
    const int kh = kc >> 2, kw = kc & 3;
    const int iy = 2 * oy - 1 + kh;
    const int ix = 2 * ox - 1 + kw;
    const bool valid = (iy >= 0) && (iy < 14) && (ix >= 0) && (ix < 14);

    _Float16* adst = Asb + arow * LSTR + aseg * 32;              // 64 B, 16B-aligned
    const _Float16* asrc = h1b + ((iy * 14 + ix) << 6) + aseg * 32;
    const _Float16* bsrc = w2row + kc * 64;                      // 128 B contig
    _Float16* bdst = Bsb + tid * LSTR;

#if USE_ASYNC_LDS
    if (valid) {
#pragma unroll
        for (int i = 0; i < 4; ++i)                              // 4 x 16 B
            async_copy_b128(adst + i * 8, asrc + i * 8);
    } else {
        uint4 z = {0u, 0u, 0u, 0u};
        uint4* zd = (uint4*)adst;
#pragma unroll
        for (int i = 0; i < 4; ++i) zd[i] = z;
    }
#pragma unroll
    for (int i = 0; i < 8; ++i)                                  // 8 x 16 B
        async_copy_b128(bdst + i * 8, bsrc + i * 8);
#else
    if (valid) {
        const uint32_t* s = (const uint32_t*)asrc;
        uint32_t* d = (uint32_t*)adst;
#pragma unroll
        for (int i = 0; i < 16; ++i) d[i] = s[i];
    } else {
        uint32_t* d = (uint32_t*)adst;
#pragma unroll
        for (int i = 0; i < 16; ++i) d[i] = 0u;
    }
    const uint32_t* s = (const uint32_t*)bsrc;
    uint32_t* d = (uint32_t*)bdst;
#pragma unroll
    for (int i = 0; i < 32; ++i) d[i] = s[i];
    if (kc + 1 < 16)   // warm next weight chunk (global_prefetch_b8)
        __builtin_prefetch(w2row + (kc + 1) * 64, 0, 0);
#endif
}

__global__ __launch_bounds__(128) void conv2_wmma(
    const _Float16* __restrict__ h1,    // [B][14][14][64]
    const _Float16* __restrict__ w2t,   // [128][1024]
    float*          __restrict__ h2)    // [M_ROWS][128]
{
    __shared__ _Float16 As[2][64 * LSTR];    // 2 x ~9 KB
    __shared__ _Float16 Bs[2][128 * LSTR];   // 2 x ~18 KB

    const int tid  = threadIdx.x;
    const int wave = tid >> 5;
    const int lane = tid & 31;
    const int lm   = lane & 15;         // M (or N) within 16-tile
    const int lg   = lane >> 4;         // half-wave group
    const int mbase = blockIdx.x * 64;

    v8f acc[8];
#pragma unroll
    for (int i = 0; i < 8; ++i)
#pragma unroll
        for (int e = 0; e < 8; ++e) acc[i][e] = 0.f;

    // Per-thread A-tile coordinates (fixed across the K loop).
    const int arow = tid >> 1;          // 0..63: tile row this thread feeds
    const int aseg = tid & 1;           // low/high 32 halves of the 64-k row
    const int gRow = mbase + arow;
    const int b_   = gRow / 49;
    const int pix  = gRow % 49;
    const int oy   = pix / 7;
    const int ox   = pix % 7;
    const _Float16* h1b   = h1 + (size_t)b_ * 12544;   // this row's sample
    const _Float16* w2row = w2t + (size_t)tid * K_DIM; // this thread's oc row

    // Prologue: stage tap 0 into buffer 0.
    stage_conv2_tiles(As[0], Bs[0], 0, tid, arow, aseg, oy, ox, h1b, w2row);

    for (int kc = 0; kc < 16; ++kc) {   // kc == tap index (kh*4+kw)
        const int cur = kc & 1;
#if USE_ASYNC_LDS
        __builtin_amdgcn_s_wait_asynccnt(0);   // this wave's stage writes landed
#endif
        __syncthreads();                        // all waves' tiles published

        // Kick off next tap into the other buffer; overlaps with compute below.
        if (kc + 1 < 16)
            stage_conv2_tiles(As[cur ^ 1], Bs[cur ^ 1], kc + 1,
                              tid, arow, aseg, oy, ox, h1b, w2row);

        // ---- compute: 2 K-sub-blocks x 8 N-tiles = 16 WMMA per wave --------
        const _Float16* arowp = &As[cur][(wave * 16 + lm) * LSTR];
#pragma unroll
        for (int ks = 0; ks < 2; ++ks) {
            union { v16h v; uint32_t u[8]; } af;
#pragma unroll
            for (int r = 0; r < 8; ++r) {
                const int kk = ks * 32 + ((r >= 4) ? 16 : 0) + lg * 8 + 2 * (r & 3);
                af.u[r] = *(const uint32_t*)&arowp[kk];
            }
#pragma unroll
            for (int nt = 0; nt < 8; ++nt) {
                union { v16h v; uint32_t u[8]; } bf;
                const _Float16* brow =
                    &Bs[cur][(nt * 16 + lm) * LSTR + ks * 32 + lg * 8];
#pragma unroll
                for (int r = 0; r < 8; ++r)
                    bf.u[r] = *(const uint32_t*)&brow[2 * (r & 3) + ((r >= 4) ? 16 : 0)];
                acc[nt] = __builtin_amdgcn_wmma_f32_16x16x32_f16(
                    false, af.v, false, bf.v, (short)0, acc[nt], false, false);
            }
        }
    }

    // ---- store: h2[row][c], rows M = base + r + 8*lg, cols N = nt*16+lm ----
    const size_t rowbase = (size_t)mbase + wave * 16;
#pragma unroll
    for (int nt = 0; nt < 8; ++nt)
#pragma unroll
        for (int r = 0; r < 8; ++r)
            h2[(rowbase + r + 8 * lg) * N_DIM + nt * 16 + lm] = acc[nt][r];
}

// ---------------------------------------------------------------------------
// Kernel 4: BN batch stats. Thread t owns channel t (coalesced rows),
// one atomic pair per block per channel.
// ---------------------------------------------------------------------------
__global__ __launch_bounds__(128) void bn_stats(
    const float* __restrict__ h2, float* __restrict__ gsum, float* __restrict__ gsq)
{
    const int c = threadIdx.x;
    float s = 0.f, q = 0.f;
    for (size_t row = blockIdx.x; row < (size_t)M_ROWS; row += gridDim.x) {
        float v = h2[row * N_DIM + c];
        s += v;
        q += v * v;
    }
    atomicAdd(&gsum[c], s);
    atomicAdd(&gsq[c], q);
}

// ---------------------------------------------------------------------------
// Kernel 5: fold mean/var/gamma/beta into per-channel scale & shift
// ---------------------------------------------------------------------------
__global__ __launch_bounds__(128) void bn_params(
    const float* __restrict__ gsum, const float* __restrict__ gsq,
    const float* __restrict__ gamma, const float* __restrict__ beta,
    float* __restrict__ scale, float* __restrict__ shift)
{
    const int c = threadIdx.x;
    const float n = (float)M_ROWS;
    float mean = gsum[c] / n;
    float var  = gsq[c] / n - mean * mean;     // biased variance
    float sc   = gamma[c] * rsqrtf(var + BN_EPS);
    scale[c] = sc;
    shift[c] = beta[c] - mean * sc;
}

// ---------------------------------------------------------------------------
// Kernel 6: BN apply + leaky + conv3 (7x7, 128ch dot) + sigmoid. Block/sample.
// ---------------------------------------------------------------------------
__global__ __launch_bounds__(128) void head(
    const float* __restrict__ h2,       // [B*49][128]
    const float* __restrict__ w3,       // [1][128][7][7]
    const float* __restrict__ scale, const float* __restrict__ shift,
    float* __restrict__ out)            // [B][1]
{
    __shared__ float ssc[128], ssh[128], red[128];
    const int b = blockIdx.x, t = threadIdx.x;
    ssc[t] = scale[t];
    ssh[t] = shift[t];
    __syncthreads();

    const float* base = h2 + (size_t)b * 49 * N_DIM;
    float acc = 0.f;
    for (int j = t; j < 49 * N_DIM; j += 128) {    // j = p*128 + c (coalesced)
        const int c = j & 127, p = j >> 7;
        float v = base[j] * ssc[c] + ssh[c];
        acc += leaky(v) * w3[c * 49 + p];
    }
    red[t] = acc;
    __syncthreads();
#pragma unroll
    for (int s = 64; s > 0; s >>= 1) {
        if (t < s) red[t] += red[t + s];
        __syncthreads();
    }
    if (t == 0) out[b] = 1.f / (1.f + __expf(-red[0]));
}

// ---------------------------------------------------------------------------
// Launcher. Workspace layout (bytes):
//   h1  f16 [8192*196*64]   205,520,896
//   w2t f16 [128*1024]          262,144
//   h2  f32 [401408*128]    205,520,896
//   gsum/gsq/scale/shift f32[128] each
// ---------------------------------------------------------------------------
extern "C" void kernel_launch(void* const* d_in, const int* in_sizes, int n_in,
                              void* d_out, int out_size, void* d_ws, size_t ws_size,
                              hipStream_t stream)
{
    const float* images = (const float*)d_in[0];
    const int*   labels = (const int*)d_in[1];
    const float* w1     = (const float*)d_in[2];
    const float* w2     = (const float*)d_in[3];
    const float* w3     = (const float*)d_in[4];
    const float* gamma  = (const float*)d_in[5];
    const float* beta   = (const float*)d_in[6];
    float* out = (float*)d_out;

    char* ws = (char*)d_ws;
    size_t off = 0;
    _Float16* h1  = (_Float16*)(ws + off); off += (size_t)B_SZ * 12544 * 2;
    _Float16* w2t = (_Float16*)(ws + off); off += (size_t)N_DIM * K_DIM * 2;
    float* h2   = (float*)(ws + off); off += (size_t)M_ROWS * N_DIM * 4;
    float* gsum = (float*)(ws + off); off += 512;
    float* gsq  = (float*)(ws + off); off += 512;
    float* scl  = (float*)(ws + off); off += 512;
    float* shf  = (float*)(ws + off); off += 512;

    zero_f32   <<<1, 256, 0, stream>>>(gsum, 256);   // gsum & gsq contiguous
    conv1_leaky<<<B_SZ, 256, 0, stream>>>(images, labels, w1, h1);
    prep_w2    <<<512, 256, 0, stream>>>(w2, w2t);
    conv2_wmma <<<M_ROWS / 64, 128, 0, stream>>>(h1, w2t, h2);
    bn_stats   <<<512, 128, 0, stream>>>(h2, gsum, gsq);
    bn_params  <<<1, 128, 0, stream>>>(gsum, gsq, gamma, beta, scl, shf);
    head       <<<B_SZ, 128, 0, stream>>>(h2, w3, scl, shf, out);
}
